// DCGRUDecoder_10471130267999
// MI455X (gfx1250) — matrix operand order repaired and checked by reading, not compiled, not run on hardware
//
#include <hip/hip_runtime.h>
#include <hip/hip_bf16.h>
#include <math.h>

typedef __attribute__((ext_vector_type(16))) _Float16 v16h;
typedef __attribute__((ext_vector_type(8)))  _Float16 v8h;
typedef __attribute__((ext_vector_type(8)))  float    v8f;
typedef __attribute__((ext_vector_type(4)))  float    f32x4;

#define NSNODE 5000
#define NSRC   8000
#define TFORE  24
#define HIDD   64
#define NWPD   32
#define ES2S   80000
#define EPATH  40000
#define NTGT   2000
#define LD0    320   // layer0 packed K (3*105 -> 320)
#define LD1    384   // layer1 packed K (3*128)
#define MTILES 313   // 5008/16

// ---------------- utility kernels ----------------
__global__ void k_fill_f32(float* p, float v, int n) {
  int i = blockIdx.x * blockDim.x + threadIdx.x;
  if (i < n) p[i] = v;
}
__global__ void k_fill_u32(unsigned* p, unsigned v, int n) {
  int i = blockIdx.x * blockDim.x + threadIdx.x;
  if (i < n) p[i] = v;
}
__global__ void k_copy_f32(float* d, const float* s, int n) {
  int i = blockIdx.x * blockDim.x + threadIdx.x;
  if (i < n) d[i] = s[i];
}
__global__ void k_copy_i32(int* d, const int* s, int n) {
  int i = blockIdx.x * blockDim.x + threadIdx.x;
  if (i < n) d[i] = s[i];
}

// ---------------- GAT (phase A) ----------------
__device__ __forceinline__ unsigned fmapu(float f) {
  unsigned u = __float_as_uint(f);
  return (u & 0x80000000u) ? ~u : (u | 0x80000000u);
}
__device__ __forceinline__ float funmap(unsigned u) {
  return __uint_as_float((u & 0x80000000u) ? (u & 0x7fffffffu) : ~u);
}

__global__ void k_gat_src(const float* __restrict__ feat, const float* __restrict__ Wsrc,
                          const float* __restrict__ attn, float* __restrict__ Xs,
                          float* __restrict__ Ssrc) {
  int idx = blockIdx.x * blockDim.x + threadIdx.x;
  if (idx >= TFORE * NSRC) return;
  const float* f = feat + (size_t)idx * 3;
  float f0 = f[0], f1 = f[1], f2 = f[2];
  float o[32];
#pragma unroll
  for (int j = 0; j < 32; ++j)
    o[j] = f0 * Wsrc[j] + f1 * Wsrc[32 + j] + f2 * Wsrc[64 + j];
  float* xo = Xs + (size_t)idx * 32;
#pragma unroll
  for (int j = 0; j < 32; ++j) xo[j] = o[j];
  float* so = Ssrc + (size_t)idx * 4;
#pragma unroll
  for (int h = 0; h < 4; ++h) {
    float s = 0.f;
#pragma unroll
    for (int d = 0; d < 8; ++d) s += o[h * 8 + d] * attn[h * 16 + d];
    so[h] = s;
  }
}

__global__ void k_gat_edge(const float* __restrict__ eattr, const float* __restrict__ Wedge,
                           const float* __restrict__ attn, float* __restrict__ ee,
                           float* __restrict__ Se) {
  int e = blockIdx.x * blockDim.x + threadIdx.x;
  if (e >= EPATH) return;
  const float* f = eattr + (size_t)e * 3;
  float f0 = f[0], f1 = f[1], f2 = f[2];
  float o[32];
#pragma unroll
  for (int j = 0; j < 32; ++j)
    o[j] = f0 * Wedge[j] + f1 * Wedge[32 + j] + f2 * Wedge[64 + j];
  float* eo = ee + (size_t)e * 32;
#pragma unroll
  for (int j = 0; j < 32; ++j) eo[j] = o[j];
  float* so = Se + (size_t)e * 4;
#pragma unroll
  for (int h = 0; h < 4; ++h) {
    float s = 0.f;
#pragma unroll
    for (int d = 0; d < 8; ++d) s += o[h * 8 + d] * attn[h * 16 + 8 + d];
    so[h] = s;
  }
}

__global__ void k_gat_max(const float* __restrict__ Ssrc, const float* __restrict__ Se,
                          const int* __restrict__ src, const int* __restrict__ dst,
                          unsigned* __restrict__ M) {
  int idx = blockIdx.x * blockDim.x + threadIdx.x;
  if (idx >= EPATH * TFORE) return;
  int e = idx % EPATH, t = idx / EPATH;
  int s = src[e], d = dst[e];
  const float* ss = Ssrc + ((size_t)t * NSRC + s) * 4;
  const float* se = Se + (size_t)e * 4;
  unsigned* mr = M + ((size_t)t * NSNODE + d) * 4;
#pragma unroll
  for (int h = 0; h < 4; ++h) {
    float v = ss[h] + se[h];
    v = v > 0.f ? v : 0.2f * v;
    atomicMax(&mr[h], fmapu(v));
  }
}

__global__ void k_gat_sum(const float* __restrict__ Ssrc, const float* __restrict__ Se,
                          const float* __restrict__ Xs, const float* __restrict__ ee,
                          const int* __restrict__ src, const int* __restrict__ dst,
                          const unsigned* __restrict__ M, float* __restrict__ den,
                          float* __restrict__ Nm) {
  int idx = blockIdx.x * blockDim.x + threadIdx.x;
  if (idx >= EPATH * TFORE) return;
  int e = idx % EPATH, t = idx / EPATH;
  int s = src[e], d = dst[e];
  const float* ss = Ssrc + ((size_t)t * NSRC + s) * 4;
  const float* se = Se + (size_t)e * 4;
  const unsigned* mr = M + ((size_t)t * NSNODE + d) * 4;
#pragma unroll
  for (int h = 0; h < 4; ++h) {
    float v = ss[h] + se[h];
    v = v > 0.f ? v : 0.2f * v;
    float ex = expf(v - funmap(mr[h]));
    atomicAdd(&den[((size_t)t * NSNODE + d) * 4 + h], ex);
    const float* xs = Xs + ((size_t)t * NSRC + s) * 32 + h * 8;
    const float* ed = ee + (size_t)e * 32 + h * 8;
    float* nm = Nm + ((size_t)t * NSNODE + d) * 32 + h * 8;
#pragma unroll
    for (int q = 0; q < 8; ++q) atomicAdd(&nm[q], (xs[q] + ed[q]) * ex);
  }
}

__global__ void k_gat_fin(float* __restrict__ nwp, const float* __restrict__ Nm,
                          const float* __restrict__ den) {
  int idx = blockIdx.x * blockDim.x + threadIdx.x;
  if (idx >= TFORE * NSNODE * NWPD) return;
  int h = (idx & 31) >> 3;
  int tn = idx >> 5;
  nwp[idx] += Nm[idx] / (den[tn * 4 + h] + 1e-9f);
}

// ---------------- CSR build (dst-sorted s2s) ----------------
__global__ void k_count_deg(const int* __restrict__ dst, int* __restrict__ deg, int E) {
  int e = blockIdx.x * blockDim.x + threadIdx.x;
  if (e < E) atomicAdd(&deg[dst[e]], 1);
}

__global__ void k_scan(const int* __restrict__ deg, int* __restrict__ rowptr, int n) {
  __shared__ int sm[1024];
  __shared__ int carry;
  int tid = threadIdx.x;
  if (tid == 0) carry = 0;
  __syncthreads();
  for (int base = 0; base < n; base += 1024) {
    int i = base + tid;
    int v = (i < n) ? deg[i] : 0;
    sm[tid] = v;
    __syncthreads();
    for (int off = 1; off < 1024; off <<= 1) {
      int t = (tid >= off) ? sm[tid - off] : 0;
      __syncthreads();
      sm[tid] += t;
      __syncthreads();
    }
    if (i < n) rowptr[i + 1] = carry + sm[tid];
    __syncthreads();
    if (tid == 0) carry += sm[1023];
    __syncthreads();
  }
  if (tid == 0) rowptr[0] = 0;
}

__global__ void k_csr_fill(const int* __restrict__ src, const int* __restrict__ dst,
                           const float* __restrict__ w, int* __restrict__ pos,
                           int* __restrict__ col, float* __restrict__ ws, int E) {
  int e = blockIdx.x * blockDim.x + threadIdx.x;
  if (e >= E) return;
  int p = atomicAdd(&pos[dst[e]], 1);
  col[p] = src[e];
  ws[p] = w[e];
}

// ---- weight pack: f32 (3,Din,Dout) -> f16 in WMMA B-fragment-swizzled order ----
// Wp[(((kcb*Ntg + ntg)*32 + lane)*16 + i] = Wcat[kcb*32 + 8*(lane>>4) + (i<8 ? i : 8+i)][ntg*16 + (lane&15)]
__global__ void k_pack_w(const float* __restrict__ W, _Float16* __restrict__ Wp,
                         int Din, int Dout, int Kpad) {
  int idx = blockIdx.x * blockDim.x + threadIdx.x;
  if (idx >= Kpad * Dout) return;
  int i = idx & 15;
  int lane = (idx >> 4) & 31;
  int rest = idx >> 9;
  int Ntg = Dout >> 4;
  int ntg = rest % Ntg;
  int kcb = rest / Ntg;
  int k = kcb * 32 + 8 * (lane >> 4) + (i < 8 ? i : 8 + i);
  int n = ntg * 16 + (lane & 15);
  float v = (k < 3 * Din) ? W[(size_t)k * Dout + n] : 0.f;  // k = hop*Din+d contiguous
  Wp[idx] = (_Float16)v;
}

// ---------------- step kernels ----------------
__global__ void k_build_x0(const float* __restrict__ y, const float* __restrict__ nwp_t,
                           const float* __restrict__ st, const float* __restrict__ h0,
                           float* __restrict__ Xr, float* __restrict__ Xc) {
  int n = blockIdx.x, c = threadIdx.x;
  if (c >= 105) return;
  float v;
  if (c == 0)      v = y[n];
  else if (c < 33) v = nwp_t[n * 32 + (c - 1)];
  else if (c < 41) v = st[n * 8 + (c - 33)];
  else             v = h0[n * 64 + (c - 41)];
  Xr[(size_t)n * LD0 + c] = v;
  if (c < 41) Xc[(size_t)n * LD0 + c] = v;
}

__global__ void k_build_x1(const float* __restrict__ h0n, const float* __restrict__ h1,
                           float* __restrict__ Xr, float* __restrict__ Xc) {
  int n = blockIdx.x, c = threadIdx.x;
  if (c >= 128) return;
  if (c < 64) {
    float v = h0n[n * 64 + c];
    Xr[(size_t)n * LD1 + c] = v;
    Xc[(size_t)n * LD1 + c] = v;
  } else {
    Xr[(size_t)n * LD1 + c] = h1[n * 64 + (c - 64)];
  }
}

// one block per dst node; thread = feature dim; CSR gather (no atomics)
__global__ void k_diffuse(const float* __restrict__ src, int so, float* __restrict__ dst,
                          int dof, float* __restrict__ dst2, int dup, int ld, int Din,
                          const int* __restrict__ rowptr, const int* __restrict__ col,
                          const float* __restrict__ w) {
  int n = blockIdx.x, d = threadIdx.x;
  if (d >= Din) return;
  int e0 = rowptr[n], e1 = rowptr[n + 1];
  float acc = 0.f;
  for (int e = e0; e < e1; ++e) {
    if (e + 1 < e1)  // prefetch next neighbor's row segment (global_prefetch_b8)
      __builtin_prefetch(&src[(size_t)col[e + 1] * ld + so + d], 0, 1);
    acc += w[e] * src[(size_t)col[e] * ld + so + d];
  }
  dst[(size_t)n * ld + dof + d] = acc;
  if (d < dup) dst2[(size_t)n * ld + dof + d] = acc;
}

// WMMA GEMM: wave computes a 16-row x 64-col strip (4 wmma per K-chunk, shared A frag).
// A frag = two contiguous 8-float runs per lane (b128 loads + cvt_pk); B frag = two
// contiguous b128 loads from fragment-swizzled weights (lane-coalesced).
// MODE 0 = RU (sigmoid; r*h -> packed c-input, u -> Ubuf); MODE 1 = C (state update).
template <int MODE>
__global__ __launch_bounds__(128) void k_wmma_gemm(
    const float* __restrict__ X, int ldx, const _Float16* __restrict__ Wsw, int Ntg,
    int K, const float* __restrict__ bias, int Mreal, int Ngrp,
    const float* __restrict__ hold, float* __restrict__ rh_dst, int rh_ld, int rh_off,
    float* __restrict__ Ubuf, float* __restrict__ hnew) {
  int wave = threadIdx.x >> 5;
  int lane = threadIdx.x & 31;
  int tile = blockIdx.x * 4 + wave;
  if (tile >= MTILES * Ngrp) return;  // uniform per wave -> EXEC all ones for WMMA
  int mt = tile % MTILES;
  int ng = tile / MTILES;
  int half = lane >> 4, l16 = lane & 15;
  int arow = mt * 16 + l16;
  v8f acc0 = {}, acc1 = {}, acc2 = {}, acc3 = {};
  int Kt = K >> 5;
  for (int kcb = 0; kcb < Kt; ++kcb) {
    const float* rp = X + (size_t)arow * ldx + kcb * 32 + 8 * half;
    f32x4 a0 = *(const f32x4*)(rp);
    f32x4 a1 = *(const f32x4*)(rp + 4);
    f32x4 a2 = *(const f32x4*)(rp + 16);
    f32x4 a3 = *(const f32x4*)(rp + 20);
    v16h a;
#pragma unroll
    for (int q = 0; q < 4; ++q) {
      a[q]      = (_Float16)a0[q];
      a[4 + q]  = (_Float16)a1[q];
      a[8 + q]  = (_Float16)a2[q];
      a[12 + q] = (_Float16)a3[q];
    }
    const _Float16* bbase = Wsw + (((size_t)kcb * Ntg + ng * 4) * 32 + lane) * 16;
#pragma unroll
    for (int nt = 0; nt < 4; ++nt) {
      v8h blo = *(const v8h*)(bbase + (size_t)nt * 512);
      v8h bhi = *(const v8h*)(bbase + (size_t)nt * 512 + 8);
      v16h b;
#pragma unroll
      for (int q = 0; q < 8; ++q) {
        b[q] = blo[q];
        b[8 + q] = bhi[q];
      }
      v8f& acc = (nt == 0) ? acc0 : (nt == 1) ? acc1 : (nt == 2) ? acc2 : acc3;
      acc = __builtin_amdgcn_wmma_f32_16x16x32_f16(false, a, false, b, (short)0, acc,
                                                   false, false);
    }
  }
  int mbase = mt * 16 + (half ? 8 : 0);
#pragma unroll
  for (int nt = 0; nt < 4; ++nt) {
    v8f acc = (nt == 0) ? acc0 : (nt == 1) ? acc1 : (nt == 2) ? acc2 : acc3;
    int gn = ng * 64 + nt * 16 + l16;
    float bv = bias[gn];
#pragma unroll
    for (int r = 0; r < 8; ++r) {
      int gm = mbase + r;
      if (gm >= Mreal) continue;
      float v = acc[r] + bv;
      if (MODE == 0) {
        float s = 1.0f / (1.0f + expf(-v));
        if (gn < HIDD)
          rh_dst[(size_t)gm * rh_ld + rh_off + gn] = s * hold[gm * HIDD + gn];
        else
          Ubuf[gm * HIDD + (gn - HIDD)] = s;
      } else {
        float c = tanhf(v);
        float u = Ubuf[gm * HIDD + gn];
        hnew[gm * HIDD + gn] = u * hold[gm * HIDD + gn] + (1.0f - u) * c;
      }
    }
  }
}

__global__ void k_out_head(const float* __restrict__ h1, const float* __restrict__ W,
                           const float* __restrict__ b, float* __restrict__ y) {
  int n = blockIdx.x * blockDim.x + threadIdx.x;
  if (n >= NSNODE) return;
  float acc = b[0];
#pragma unroll
  for (int k = 0; k < 64; ++k) acc += h1[n * 64 + k] * W[k];
  y[n] = acc;
}

__global__ void k_gather(const float* __restrict__ y, const int* __restrict__ mask,
                         float* __restrict__ out, int t) {
  int j = blockIdx.x * blockDim.x + threadIdx.x;
  if (j >= NTGT) return;
  out[j * TFORE + t] = y[mask[j]];
}

// ---------------- host orchestration ----------------
static inline int cdiv(int a, int b) { return (a + b - 1) / b; }

extern "C" void kernel_launch(void* const* d_in, const int* in_sizes, int n_in,
                              void* d_out, int out_size, void* d_ws, size_t ws_size,
                              hipStream_t stream) {
  (void)in_sizes; (void)n_in; (void)out_size; (void)ws_size;
  const float* H_init = (const float*)d_in[0];
  const float* icond  = (const float*)d_in[1];
  const float* ecmwf  = (const float*)d_in[2];
  const float* statp  = (const float*)d_in[3];
  const int*   s2s    = (const int*)d_in[4];
  const float* s2sw   = (const float*)d_in[5];
  const int*   i2s    = (const int*)d_in[6];
  const float* i2sa   = (const float*)d_in[7];
  const int*   e2s    = (const int*)d_in[8];
  const float* e2sa   = (const float*)d_in[9];
  const float* ylast  = (const float*)d_in[10];
  const int*   tmask  = (const int*)d_in[11];
  const float* Wsrc_i = (const float*)d_in[12];
  const float* Wedge_i= (const float*)d_in[13];
  const float* attn_i = (const float*)d_in[14];
  const float* Wsrc_e = (const float*)d_in[15];
  const float* Wedge_e= (const float*)d_in[16];
  const float* attn_e = (const float*)d_in[17];
  const float* W_ru0  = (const float*)d_in[18];
  const float* b_ru0  = (const float*)d_in[19];
  const float* W_c0   = (const float*)d_in[20];
  const float* b_c0   = (const float*)d_in[21];
  const float* W_ru1  = (const float*)d_in[22];
  const float* b_ru1  = (const float*)d_in[23];
  const float* W_c1   = (const float*)d_in[24];
  const float* b_c1   = (const float*)d_in[25];
  const float* outW   = (const float*)d_in[26];
  const float* outb   = (const float*)d_in[27];
  float* out = (float*)d_out;

  // ---- workspace bump allocator ----
  char* base = (char*)d_ws;
  size_t off = 0;
  auto alloc = [&](size_t bytes) -> void* {
    void* p = base + off;
    off = (off + bytes + 255) & ~(size_t)255;
    return p;
  };
  // persistent
  float* nwp   = (float*)alloc((size_t)TFORE * NSNODE * NWPD * 4);
  int*   rowptr= (int*)alloc((NSNODE + 1) * 4);
  int*   colIdx= (int*)alloc(ES2S * 4);
  float* wS    = (float*)alloc(ES2S * 4);
  int*   deg   = (int*)alloc(NSNODE * 4);  // reused as fill cursor
  float* h0a   = (float*)alloc(NSNODE * HIDD * 4);
  float* h0b   = (float*)alloc(NSNODE * HIDD * 4);
  float* h1a   = (float*)alloc(NSNODE * HIDD * 4);
  float* h1b   = (float*)alloc(NSNODE * HIDD * 4);
  float* ybuf  = (float*)alloc(NSNODE * 4);
  _Float16* Wru0p = (_Float16*)alloc((size_t)LD0 * 128 * 2);
  _Float16* Wc0p  = (_Float16*)alloc((size_t)LD0 * 64 * 2);
  _Float16* Wru1p = (_Float16*)alloc((size_t)LD1 * 128 * 2);
  _Float16* Wc1p  = (_Float16*)alloc((size_t)LD1 * 64 * 2);
  // scratch union: phase-A view and step view share memory
  size_t scratch0 = off;
  float*    Xs   = (float*)alloc((size_t)TFORE * NSRC * 32 * 4);
  float*    Ssrc = (float*)alloc((size_t)TFORE * NSRC * 4 * 4);
  float*    eeB  = (float*)alloc((size_t)EPATH * 32 * 4);
  float*    SeB  = (float*)alloc((size_t)EPATH * 4 * 4);
  unsigned* Mb   = (unsigned*)alloc((size_t)TFORE * NSNODE * 4 * 4);
  float*    den  = (float*)alloc((size_t)TFORE * NSNODE * 4 * 4);
  float*    Nm   = (float*)alloc((size_t)TFORE * NSNODE * 32 * 4);
  off = scratch0;  // alias step buffers over phase-A scratch (phase A finishes first)
  float* Xr0 = (float*)alloc((size_t)5008 * LD0 * 4);
  float* Xc0 = (float*)alloc((size_t)5008 * LD0 * 4);
  float* Xr1 = (float*)alloc((size_t)5008 * LD1 * 4);
  float* Xc1 = (float*)alloc((size_t)5008 * LD1 * 4);
  float* U0  = (float*)alloc((size_t)NSNODE * HIDD * 4);
  float* U1  = (float*)alloc((size_t)NSNODE * HIDD * 4);

  const int TB = 256;

  // ---- Phase A: GAT attention fusion -> nwp (24,5000,32) ----
  k_fill_f32<<<cdiv(TFORE * NSNODE * NWPD, TB), TB, 0, stream>>>(nwp, 0.f,
                                                                 TFORE * NSNODE * NWPD);
  for (int path = 0; path < 2; ++path) {
    const float* feat  = path ? ecmwf : icond;
    const float* eattr = path ? e2sa : i2sa;
    const int*   eidx  = path ? e2s : i2s;
    const float* Wsrc  = path ? Wsrc_e : Wsrc_i;
    const float* Wedge = path ? Wedge_e : Wedge_i;
    const float* attn  = path ? attn_e : attn_i;
    k_gat_src<<<cdiv(TFORE * NSRC, TB), TB, 0, stream>>>(feat, Wsrc, attn, Xs, Ssrc);
    k_gat_edge<<<cdiv(EPATH, TB), TB, 0, stream>>>(eattr, Wedge, attn, eeB, SeB);
    k_fill_u32<<<cdiv(TFORE * NSNODE * 4, TB), TB, 0, stream>>>(Mb, 0u,
                                                                TFORE * NSNODE * 4);
    k_gat_max<<<cdiv(EPATH * TFORE, TB), TB, 0, stream>>>(Ssrc, SeB, eidx, eidx + EPATH,
                                                          Mb);
    k_fill_f32<<<cdiv(TFORE * NSNODE * 4, TB), TB, 0, stream>>>(den, 0.f,
                                                                TFORE * NSNODE * 4);
    k_fill_f32<<<cdiv(TFORE * NSNODE * 32, TB), TB, 0, stream>>>(Nm, 0.f,
                                                                 TFORE * NSNODE * 32);
    k_gat_sum<<<cdiv(EPATH * TFORE, TB), TB, 0, stream>>>(Ssrc, SeB, Xs, eeB, eidx,
                                                          eidx + EPATH, Mb, den, Nm);
    k_gat_fin<<<cdiv(TFORE * NSNODE * 32, TB), TB, 0, stream>>>(nwp, Nm, den);
  }

  // ---- CSR build for s2s (dst-sorted) ----
  k_fill_u32<<<cdiv(NSNODE, TB), TB, 0, stream>>>((unsigned*)deg, 0u, NSNODE);
  k_count_deg<<<cdiv(ES2S, TB), TB, 0, stream>>>(s2s + ES2S, deg, ES2S);
  k_scan<<<1, 1024, 0, stream>>>(deg, rowptr, NSNODE);
  k_copy_i32<<<cdiv(NSNODE, TB), TB, 0, stream>>>(deg, rowptr, NSNODE);
  k_csr_fill<<<cdiv(ES2S, TB), TB, 0, stream>>>(s2s, s2s + ES2S, s2sw, deg, colIdx, wS,
                                                ES2S);

  // ---- weights -> f16 fragment-swizzled ----
  k_pack_w<<<cdiv(LD0 * 128, TB), TB, 0, stream>>>(W_ru0, Wru0p, 105, 128, LD0);
  k_pack_w<<<cdiv(LD0 * 64, TB), TB, 0, stream>>>(W_c0, Wc0p, 105, 64, LD0);
  k_pack_w<<<cdiv(LD1 * 128, TB), TB, 0, stream>>>(W_ru1, Wru1p, 128, 128, LD1);
  k_pack_w<<<cdiv(LD1 * 64, TB), TB, 0, stream>>>(W_c1, Wc1p, 128, 64, LD1);

  // ---- init states + zero packed matrices (pad rows/cols stay zero) ----
  k_copy_f32<<<cdiv(NSNODE * HIDD, TB), TB, 0, stream>>>(h0a, H_init, NSNODE * HIDD);
  k_copy_f32<<<cdiv(NSNODE * HIDD, TB), TB, 0, stream>>>(h1a, H_init + NSNODE * HIDD,
                                                         NSNODE * HIDD);
  k_copy_f32<<<cdiv(NSNODE, TB), TB, 0, stream>>>(ybuf, ylast, NSNODE);
  k_fill_f32<<<cdiv(5008 * LD0, TB), TB, 0, stream>>>(Xr0, 0.f, 5008 * LD0);
  k_fill_f32<<<cdiv(5008 * LD0, TB), TB, 0, stream>>>(Xc0, 0.f, 5008 * LD0);
  k_fill_f32<<<cdiv(5008 * LD1, TB), TB, 0, stream>>>(Xr1, 0.f, 5008 * LD1);
  k_fill_f32<<<cdiv(5008 * LD1, TB), TB, 0, stream>>>(Xc1, 0.f, 5008 * LD1);

  // ---- sequential scan over T ----
  for (int t = 0; t < TFORE; ++t) {
    float* h0c = (t & 1) ? h0b : h0a;
    float* h0n = (t & 1) ? h0a : h0b;
    float* h1c = (t & 1) ? h1b : h1a;
    float* h1n = (t & 1) ? h1a : h1b;
    const float* nwp_t = nwp + (size_t)t * NSNODE * NWPD;

    // layer 0
    k_build_x0<<<NSNODE, 128, 0, stream>>>(ybuf, nwp_t, statp, h0c, Xr0, Xc0);
    k_diffuse<<<NSNODE, 128, 0, stream>>>(Xr0, 0, Xr0, 105, Xc0, 41, LD0, 105, rowptr,
                                          colIdx, wS);
    k_diffuse<<<NSNODE, 128, 0, stream>>>(Xr0, 105, Xr0, 210, Xc0, 41, LD0, 105, rowptr,
                                          colIdx, wS);
    k_wmma_gemm<0><<<cdiv(MTILES * 2, 4), 128, 0, stream>>>(
        Xr0, LD0, Wru0p, 8, LD0, b_ru0, NSNODE, 2, h0c, Xc0, LD0, 41, U0, nullptr);
    k_diffuse<<<NSNODE, 128, 0, stream>>>(Xc0, 41, Xc0, 146, Xc0, 0, LD0, 64, rowptr,
                                          colIdx, wS);
    k_diffuse<<<NSNODE, 128, 0, stream>>>(Xc0, 146, Xc0, 251, Xc0, 0, LD0, 64, rowptr,
                                          colIdx, wS);
    k_wmma_gemm<1><<<cdiv(MTILES * 1, 4), 128, 0, stream>>>(
        Xc0, LD0, Wc0p, 4, LD0, b_c0, NSNODE, 1, h0c, nullptr, 0, 0, U0, h0n);

    // layer 1
    k_build_x1<<<NSNODE, 128, 0, stream>>>(h0n, h1c, Xr1, Xc1);
    k_diffuse<<<NSNODE, 128, 0, stream>>>(Xr1, 0, Xr1, 128, Xc1, 64, LD1, 128, rowptr,
                                          colIdx, wS);
    k_diffuse<<<NSNODE, 128, 0, stream>>>(Xr1, 128, Xr1, 256, Xc1, 64, LD1, 128, rowptr,
                                          colIdx, wS);
    k_wmma_gemm<0><<<cdiv(MTILES * 2, 4), 128, 0, stream>>>(
        Xr1, LD1, Wru1p, 8, LD1, b_ru1, NSNODE, 2, h1c, Xc1, LD1, 64, U1, nullptr);
    k_diffuse<<<NSNODE, 128, 0, stream>>>(Xc1, 64, Xc1, 192, Xc1, 0, LD1, 64, rowptr,
                                          colIdx, wS);
    k_diffuse<<<NSNODE, 128, 0, stream>>>(Xc1, 192, Xc1, 320, Xc1, 0, LD1, 64, rowptr,
                                          colIdx, wS);
    k_wmma_gemm<1><<<cdiv(MTILES * 1, 4), 128, 0, stream>>>(
        Xc1, LD1, Wc1p, 4, LD1, b_c1, NSNODE, 1, h1c, nullptr, 0, 0, U1, h1n);

    // output head + gather
    k_out_head<<<cdiv(NSNODE, TB), TB, 0, stream>>>(h1n, outW, outb, ybuf);
    k_gather<<<cdiv(NTGT, TB), TB, 0, stream>>>(ybuf, tmask, out, t);
  }
}
